// TransformerXL_36670430773895
// MI455X (gfx1250) — matrix operand order repaired and checked by reading, not compiled, not run on hardware
//
#include <hip/hip_runtime.h>
#include <hip/hip_bf16.h>
#include <cstdint>
#include <cstddef>

// ---------------------------------------------------------------------------
// Types for CDNA5 WMMA (wave32, 16x16x32 bf16 -> f32)
// ---------------------------------------------------------------------------
typedef __attribute__((ext_vector_type(16))) __bf16 v16bf;
typedef __attribute__((ext_vector_type(8)))  float  v8f;

union FragBF { unsigned int u[8]; v16bf v; };

#define DEV __device__ __forceinline__

constexpr int   Bc = 4, Nc = 1024, Dc = 1024, Hc = 16, DHc = 64;
constexpr int   FIc = 2730, FI2c = 5460;
constexpr int   FIpc = 2752;                     // FI padded to mult. of 32
constexpr int   Mrows = Bc * Nc;                 // 4096 tokens (mult. of 128)
constexpr float EPSc  = 1.1920929e-07f;

DEV unsigned short f32_bf16(float f) {
  unsigned int u = __float_as_uint(f);
  u += 0x7fffu + ((u >> 16) & 1u);               // round-to-nearest-even
  return (unsigned short)(u >> 16);
}
DEV v8f v8f_zero() {
  v8f z;
#pragma unroll
  for (int i = 0; i < 8; ++i) z[i] = 0.f;
  return z;
}
DEV float sigmoidf_(float x) { return 1.f / (1.f + __expf(-x)); }

// --- DPP16 cross-lane helpers (VALU, no LDS): row_ror:n = 0x120+n ----------
template <int CTRL>
DEV float dpp_f(float v) {
  return __int_as_float(__builtin_amdgcn_update_dpp(
      0, __float_as_int(v), CTRL, 0xf, 0xf, true));
}
// Reduce over each 16-lane row via rotations 1,2,4,8 (all lanes get result).
DEV float rowmax16(float v) {
  v = fmaxf(v, dpp_f<0x121>(v));
  v = fmaxf(v, dpp_f<0x122>(v));
  v = fmaxf(v, dpp_f<0x124>(v));
  v = fmaxf(v, dpp_f<0x128>(v));
  return v;
}
DEV float rowsum16(float v) {
  v += dpp_f<0x121>(v);
  v += dpp_f<0x122>(v);
  v += dpp_f<0x124>(v);
  v += dpp_f<0x128>(v);
  return v;
}
// Swap adjacent lanes (t ^ 1): quad_perm(1,0,3,2) = 0xB1
DEV float pairswap(float v) { return dpp_f<0xB1>(v); }
// Full wave32 sum: DPP within 16-lane rows, one bpermute across halves.
DEV float wavesum32(float v) {
  v = rowsum16(v);
  v += __shfl_xor(v, 16);
  return v;
}

// ---------------------------------------------------------------------------
// fp32 -> bf16 casts (weights, once per call)
// ---------------------------------------------------------------------------
__global__ void cast_f32_bf16(const float* __restrict__ s,
                              unsigned short* __restrict__ d, int n) {
  for (int i = blockIdx.x * blockDim.x + threadIdx.x; i < n;
       i += gridDim.x * blockDim.x)
    d[i] = f32_bf16(s[i]);
}

// Cast with K-padding: dst[r*Kp + k] = (k<K) ? bf16(src[r*K+k]) : 0
__global__ void cast_f32_bf16_pad(const float* __restrict__ s,
                                  unsigned short* __restrict__ d,
                                  int rows, int K, int Kp) {
  const int total = rows * Kp;
  for (int i = blockIdx.x * blockDim.x + threadIdx.x; i < total;
       i += gridDim.x * blockDim.x) {
    const int r = i / Kp, k = i - r * Kp;
    d[i] = (k < K) ? f32_bf16(s[(size_t)r * K + k]) : (unsigned short)0;
  }
}

// ---------------------------------------------------------------------------
// Fused (double-)RMSNorm, bf16 output. w2 == nullptr -> single norm.
// ---------------------------------------------------------------------------
__global__ __launch_bounds__(256) void rmsnorm2_bf16(
    const float* __restrict__ x, const float* __restrict__ w1,
    const float* __restrict__ w2, unsigned short* __restrict__ out) {
  __shared__ float red1[8], red2[8];
  const int row = blockIdx.x, tid = threadIdx.x;
  const int lane = tid & 31, wave = tid >> 5;
  const float4 xv = ((const float4*)(x + (size_t)row * Dc))[tid];
  float ss = xv.x * xv.x + xv.y * xv.y + xv.z * xv.z + xv.w * xv.w;
  ss = wavesum32(ss);
  if (lane == 0) red1[wave] = ss;
  __syncthreads();
  float tot = 0.f;
#pragma unroll
  for (int i = 0; i < 8; ++i) tot += red1[i];
  const float r1 = rsqrtf(tot * (1.f / Dc) + EPSc);
  const float4 w1v = ((const float4*)w1)[tid];
  float y0 = xv.x * r1 * w1v.x, y1 = xv.y * r1 * w1v.y;
  float y2 = xv.z * r1 * w1v.z, y3 = xv.w * r1 * w1v.w;
  if (w2) {
    float ss2 = y0 * y0 + y1 * y1 + y2 * y2 + y3 * y3;
    ss2 = wavesum32(ss2);
    if (lane == 0) red2[wave] = ss2;
    __syncthreads();
    float tot2 = 0.f;
#pragma unroll
    for (int i = 0; i < 8; ++i) tot2 += red2[i];
    const float r2 = rsqrtf(tot2 * (1.f / Dc) + EPSc);
    const float4 w2v = ((const float4*)w2)[tid];
    y0 *= r2 * w2v.x; y1 *= r2 * w2v.y; y2 *= r2 * w2v.z; y3 *= r2 * w2v.w;
  }
  unsigned short* o = out + (size_t)row * Dc + tid * 4;
  o[0] = f32_bf16(y0); o[1] = f32_bf16(y1);
  o[2] = f32_bf16(y2); o[3] = f32_bf16(y3);
}

// Final RMSNorm, fp32 output.
__global__ __launch_bounds__(256) void final_rms(
    const float* __restrict__ x, const float* __restrict__ w,
    float* __restrict__ out) {
  __shared__ float red[8];
  const int row = blockIdx.x, tid = threadIdx.x;
  const int lane = tid & 31, wave = tid >> 5;
  const float4 xv = ((const float4*)(x + (size_t)row * Dc))[tid];
  float ss = xv.x * xv.x + xv.y * xv.y + xv.z * xv.z + xv.w * xv.w;
  ss = wavesum32(ss);
  if (lane == 0) red[wave] = ss;
  __syncthreads();
  float tot = 0.f;
#pragma unroll
  for (int i = 0; i < 8; ++i) tot += red[i];
  const float r1 = rsqrtf(tot * (1.f / Dc) + EPSc);
  const float4 wv = ((const float4*)w)[tid];
  float4 o;
  o.x = xv.x * r1 * wv.x; o.y = xv.y * r1 * wv.y;
  o.z = xv.z * r1 * wv.z; o.w = xv.w * r1 * wv.w;
  ((float4*)(out + (size_t)row * Dc))[tid] = o;
}

// ---------------------------------------------------------------------------
// Tiled GEMM:  C[M,N] (+)= A[M,K](bf16,lda) * W[N,K]^T(bf16,ldw)  (+ bias[n])
// Preconditions (by construction): M % 128 == 0, K % 32 == 0,
// lda/ldw % 8 == 0 (rows 16B aligned). Only N may have a tail.
// Block tile 128x64, BK=32, 8 wave32 waves (2x2 WMMA each), double-buffered
// LDS with register prefetch pipeline: one barrier per K-step.
// ---------------------------------------------------------------------------
template <int ACCUM, int BIAS>
__global__ __launch_bounds__(256) void gemm_bf16_wmma(
    const unsigned short* __restrict__ A, int lda,
    const unsigned short* __restrict__ W, int ldw,
    float* __restrict__ C, const float* __restrict__ bias,
    int M, int N, int K) {
  __shared__ __align__(16) unsigned short sA[2][128 * 40];  // [m][k] pad 40
  __shared__ __align__(16) unsigned short sW[2][64 * 40];   // [n][k] pad 40

  const int tid = threadIdx.x;
  const int wave = tid >> 5, lane = tid & 31;
  const int hi = lane >> 4, ln = lane & 15;
  const int waveM = wave >> 1, waveN = wave & 1;
  const int bm = blockIdx.y * 128, bn = blockIdx.x * 64;

  // staging map: A = 512 uint4 chunks (2/thread), W = 256 uint4 (1/thread)
  const int arow = tid >> 2, acol = (tid & 3) << 3;  // rows arow, arow+64
  const int wrow = tid >> 2, wcol = (tid & 3) << 3;
  const int gnW = min(bn + wrow, N - 1);             // clamp N tail (no zero
                                                     // needed: col n only
                                                     // reads W row n)
  const unsigned short* Ap0 = A + (size_t)(bm + arow) * lda + acol;
  const unsigned short* Ap1 = A + (size_t)(bm + arow + 64) * lda + acol;
  const unsigned short* Wp  = W + (size_t)gnW * ldw + wcol;

  uint4 ra0, ra1, rw;
  auto fetch = [&](int k0) {
    ra0 = *(const uint4*)(Ap0 + k0);
    ra1 = *(const uint4*)(Ap1 + k0);
    rw  = *(const uint4*)(Wp + k0);
  };
  auto stage = [&](int buf) {
    *(uint4*)&sA[buf][arow * 40 + acol]        = ra0;
    *(uint4*)&sA[buf][(arow + 64) * 40 + acol] = ra1;
    *(uint4*)&sW[buf][wrow * 40 + wcol]        = rw;
  };

  v8f acc[2][2];
#pragma unroll
  for (int tm = 0; tm < 2; ++tm)
#pragma unroll
    for (int tn = 0; tn < 2; ++tn) acc[tm][tn] = v8f_zero();

  const int nk = K >> 5;
  fetch(0);
  stage(0);
  __syncthreads();

  for (int kt = 0; kt < nk; ++kt) {
    const int buf = kt & 1;
    const bool more = (kt + 1) < nk;
    if (more) fetch((kt + 1) << 5);  // overlap HBM with WMMA below

    FragBF aF[2], bF[2];
#pragma unroll
    for (int tm = 0; tm < 2; ++tm) {
      const int mr = waveM * 32 + tm * 16 + ln;
#pragma unroll
      for (int p = 0; p < 8; ++p) {
        const int kb = (p < 4) ? (hi * 8 + (p << 1))
                               : (16 + hi * 8 + ((p - 4) << 1));
        aF[tm].u[p] = *(const unsigned int*)&sA[buf][mr * 40 + kb];
      }
    }
#pragma unroll
    for (int tn = 0; tn < 2; ++tn) {
      const int nr = waveN * 32 + tn * 16 + ln;
#pragma unroll
      for (int p = 0; p < 8; ++p)
        bF[tn].u[p] =
            *(const unsigned int*)&sW[buf][nr * 40 + ((p < 4) ? (hi * 8 + (p << 1)) : (16 + hi * 8 + ((p - 4) << 1)))];
    }
#pragma unroll
    for (int tm = 0; tm < 2; ++tm)
#pragma unroll
      for (int tn = 0; tn < 2; ++tn)
        acc[tm][tn] = __builtin_amdgcn_wmma_f32_16x16x32_bf16(
            false, aF[tm].v, false, bF[tn].v, (short)0, acc[tm][tn],
            false, false);

    if (more) stage(buf ^ 1);
    __syncthreads();
  }

  // epilogue (m always < M by construction; only N can have a tail)
#pragma unroll
  for (int tn = 0; tn < 2; ++tn) {
    const int n = bn + waveN * 32 + tn * 16 + ln;
    if (n < N) {
      float bv = 0.f;
      if (BIAS) bv = bias[n];
#pragma unroll
      for (int tm = 0; tm < 2; ++tm) {
        const int mb = bm + waveM * 32 + tm * 16 + hi * 8;
#pragma unroll
        for (int r = 0; r < 8; ++r) {
          const size_t idx = (size_t)(mb + r) * N + n;
          const float v = acc[tm][tn][r] + bv;
          if (ACCUM) C[idx] += v; else C[idx] = v;
        }
      }
    }
  }
}

// ---------------------------------------------------------------------------
// Fused RoPE(+scale) for q,k + value-residual lerp, fp32 -> bf16 [B,H,N,DH]
// ---------------------------------------------------------------------------
__global__ __launch_bounds__(64) void prep_qkv(
    const float* __restrict__ qf, const float* __restrict__ kvf,
    const float* __restrict__ mixb, float* __restrict__ vres,
    unsigned short* __restrict__ qb, unsigned short* __restrict__ kb,
    unsigned short* __restrict__ vb, int isLayer0) {
  const int n = blockIdx.x, h = blockIdx.y, b = blockIdx.z;
  const int t = threadIdx.x;
  const int p = t >> 1;
  const float inv = __expf(-0.28782313f * (float)p);  // 10000^(-2p/64)
  float sn, cs;
  __sincosf((float)n * inv, &sn, &cs);

  const size_t tok = (size_t)b * Nc + n;
  const float qv = qf[tok * (Hc * DHc) + h * DHc + t] * 0.125f;  // DH^-0.5
  const float kv = kvf[tok * (2 * Hc * DHc) + h * DHc + t];
  const float vv = kvf[tok * (2 * Hc * DHc) + Hc * DHc + h * DHc + t];

  const float qo = pairswap(qv);  // DPP quad_perm(1,0,3,2): partner t^1
  const float ko = pairswap(kv);
  const float qrot = (t & 1) ? qo : -qo;
  const float krot = (t & 1) ? ko : -ko;

  const size_t hOff = (((size_t)b * Hc + h) * Nc + n) * DHc + t;
  qb[hOff] = f32_bf16(qv * cs + qrot * sn);
  kb[hOff] = f32_bf16(kv * cs + krot * sn);

  float vout = vv;
  if (isLayer0) {
    vres[hOff] = vv;                       // capture value residual (layer 0 v)
  } else {
    const float mix = sigmoidf_(mixb[tok * Hc + h]);
    vout = vv + mix * (vres[hOff] - vv);   // lerp(v, value_residual, mix)
  }
  vb[hOff] = f32_bf16(vout);
}

// ---------------------------------------------------------------------------
// Causal flash attention per (b,h): WMMA for S=QK^T and O+=PV, online
// softmax (DPP row reductions), double-buffered K/V staging, gate fused.
// Block = 256 threads (8 waves), each wave owns 16 query rows, key tiles = 32.
// ---------------------------------------------------------------------------
__global__ __launch_bounds__(256) void flash_attn(
    const unsigned short* __restrict__ Q, const unsigned short* __restrict__ Kg,
    const unsigned short* __restrict__ Vg, const float* __restrict__ gateb,
    unsigned short* __restrict__ Obf) {
  __shared__ __align__(16) unsigned short sK[2][32 * 72];    // [key][hd]
  __shared__ __align__(16) unsigned short sV[2][64 * 40];    // [hd][key] (T)
  __shared__ __align__(16) unsigned short sP[8][16 * 40];    // per-wave P

  const int tid = threadIdx.x, wave = tid >> 5, lane = tid & 31;
  const int hi = lane >> 4, ln = lane & 15;
  const int h = blockIdx.y, b = blockIdx.z;
  const int qbase = blockIdx.x * 128 + wave * 16;
  const size_t headOff = ((size_t)b * Hc + h) * Nc;

  // Q fragments: 16 rows x 64 hd = two 16x32 A-frags, loaded once from HBM
  FragBF qF[2];
  {
    const size_t rowOff = (headOff + qbase + ln) * DHc;
#pragma unroll
    for (int kc = 0; kc < 2; ++kc)
#pragma unroll
      for (int p = 0; p < 8; ++p) {
        const int kb = (p < 4) ? (hi * 8 + (p << 1))
                               : (16 + hi * 8 + ((p - 4) << 1));
        qF[kc].u[p] = *(const unsigned int*)&Q[rowOff + kc * 32 + kb];
      }
  }

  // staging map: 32 keys x 64 hd = 256 uint4 (1 per thread for K and V)
  const int srow = tid >> 3, scol = (tid & 7) << 3;
  uint4 rk, rv;
  auto fetch = [&](int t) {
    const size_t gOff = (headOff + t * 32 + srow) * DHc + scol;
    rk = *(const uint4*)&Kg[gOff];
    rv = *(const uint4*)&Vg[gOff];
  };
  auto stage = [&](int buf) {
    *(uint4*)&sK[buf][srow * 72 + scol] = rk;
    const unsigned int* vp = (const unsigned int*)&rv;
#pragma unroll
    for (int j = 0; j < 4; ++j) {  // transpose V into [hd][key]
      const int hd = scol + (j << 1);
      sV[buf][hd * 40 + srow]       = (unsigned short)(vp[j] & 0xffffu);
      sV[buf][(hd + 1) * 40 + srow] = (unsigned short)(vp[j] >> 16);
    }
  };

  v8f acc[4];
#pragma unroll
  for (int t = 0; t < 4; ++t) acc[t] = v8f_zero();
  float mrow[8], lrow[8];
#pragma unroll
  for (int r = 0; r < 8; ++r) { mrow[r] = -3.0e38f; lrow[r] = 0.f; }

  const int ntiles = (blockIdx.x + 1) * 4;  // keys 0 .. (qchunk+1)*128
  fetch(0);
  stage(0);
  __syncthreads();

  for (int t = 0; t < ntiles; ++t) {
    const int kstart = t * 32;
    const int buf = t & 1;
    const bool more = (t + 1) < ntiles;
    if (more) fetch(t + 1);

    if (kstart <= qbase + 15) {  // wave-uniform: has unmasked work
      // S = Q K^T : two 16x16 C tiles over 32 keys
      v8f s[2];
#pragma unroll
      for (int tn = 0; tn < 2; ++tn) {
        s[tn] = v8f_zero();
        const int key = tn * 16 + ln;
        FragBF bK;
#pragma unroll
        for (int kc = 0; kc < 2; ++kc) {
#pragma unroll
          for (int p = 0; p < 8; ++p)
            bK.u[p] = *(const unsigned int*)
                &sK[buf][key * 72 + kc * 32 + ((p < 4) ? (hi * 8 + (p << 1)) : (16 + hi * 8 + ((p - 4) << 1)))];
          s[tn] = __builtin_amdgcn_wmma_f32_16x16x32_bf16(
              false, qF[kc].v, false, bK.v, (short)0, s[tn], false, false);
        }
      }
      // causal mask + online softmax; DPP row reductions (no LDS traffic)
#pragma unroll
      for (int r = 0; r < 8; ++r) {
        const int qi = qbase + hi * 8 + r;
        float s0 = (kstart + ln > qi)      ? -1.0e30f : s[0][r];
        float s1 = (kstart + 16 + ln > qi) ? -1.0e30f : s[1][r];
        const int prow = (hi * 8 + r) * 40;
        if (qi >= kstart) {  // uniform within each 16-lane group
          const float smax = rowmax16(fmaxf(s0, s1));
          const float mnew = fmaxf(mrow[r], smax);
          const float p0 = __expf(s0 - mnew), p1 = __expf(s1 - mnew);
          const float rs = rowsum16(p0 + p1);
          const float alpha = __expf(mrow[r] - mnew);
          lrow[r] = lrow[r] * alpha + rs;
          mrow[r] = mnew;
#pragma unroll
          for (int tt = 0; tt < 4; ++tt) acc[tt][r] *= alpha;
          sP[wave][prow + ln]      = f32_bf16(p0);
          sP[wave][prow + 16 + ln] = f32_bf16(p1);
        } else {  // row fully masked in this tile
          sP[wave][prow + ln] = 0;
          sP[wave][prow + 16 + ln] = 0;
        }
      }
      // O += P V : P as 16x32 A-frag (intra-wave LDS roundtrip), V^T B-frags
      FragBF pF;
#pragma unroll
      for (int p = 0; p < 8; ++p) {
        const int kb = (p < 4) ? (hi * 8 + (p << 1))
                               : (16 + hi * 8 + ((p - 4) << 1));
        pF.u[p] = *(const unsigned int*)&sP[wave][ln * 40 + kb];
      }
#pragma unroll
      for (int tt = 0; tt < 4; ++tt) {
        const int hd = tt * 16 + ln;
        FragBF bV;
#pragma unroll
        for (int p = 0; p < 8; ++p)
          bV.u[p] = *(const unsigned int*)
              &sV[buf][hd * 40 + ((p < 4) ? (hi * 8 + (p << 1)) : (16 + hi * 8 + ((p - 4) << 1)))];
        acc[tt] = __builtin_amdgcn_wmma_f32_16x16x32_bf16(
            false, pF.v, false, bV.v, (short)0, acc[tt], false, false);
      }
    }

    if (more) stage(buf ^ 1);
    __syncthreads();
  }

  // epilogue: 1/l normalize, gate, store bf16 in [B,N,H*DH]
#pragma unroll
  for (int r = 0; r < 8; ++r) {
    const int qi = qbase + hi * 8 + r;
    const float g = sigmoidf_(gateb[((size_t)b * Nc + qi) * Hc + h]);
    const float invl = g / lrow[r];
    const size_t o = ((size_t)b * Nc + qi) * (Hc * DHc) + h * DHc;
#pragma unroll
    for (int tt = 0; tt < 4; ++tt)
      Obf[o + tt * 16 + ln] = f32_bf16(acc[tt][r] * invl);
  }
}

// ---------------------------------------------------------------------------
// GEGLU: out_bf16[m, f] = a * 0.5*g*(1+erf(g/sqrt(2))), stride FIpc (padded,
// pad region written 0 every call so the padded GEMM K-loop sees zeros).
// ---------------------------------------------------------------------------
__global__ void geglu_kernel(const float* __restrict__ u,
                             unsigned short* __restrict__ out, int total) {
  for (int i = blockIdx.x * blockDim.x + threadIdx.x; i < total;
       i += gridDim.x * blockDim.x) {
    const int m = i / FIpc, f = i - m * FIpc;
    float r = 0.f;
    if (f < FIc) {
      const float a  = u[(size_t)m * FI2c + f];
      const float gt = u[(size_t)m * FI2c + FIc + f];
      r = a * 0.5f * gt * (1.f + erff(gt * 0.7071067811865475f));
    }
    out[i] = f32_bf16(r);
  }
}

// ---------------------------------------------------------------------------
// Host orchestration
// ---------------------------------------------------------------------------
extern "C" void kernel_launch(void* const* d_in, const int* in_sizes, int n_in,
                              void* d_out, int out_size, void* d_ws,
                              size_t ws_size, hipStream_t stream) {
  (void)in_sizes; (void)n_in; (void)out_size; (void)ws_size;

  const float* x_in   = (const float*)d_in[0];
  const float* naw    = (const float*)d_in[1];   // norm_attn_w [4,1024]
  const float* wq     = (const float*)d_in[2];   // [4,1024,1024]
  const float* wkv    = (const float*)d_in[3];   // [4,2048,1024]
  const float* wo     = (const float*)d_in[4];   // [4,1024,1024]
  const float* wg     = (const float*)d_in[5];   // [4,16,1024]
  const float* wmix   = (const float*)d_in[6];   // [4,16,1024]
  const float* nfw    = (const float*)d_in[7];   // norm_ff_w
  const float* ffw    = (const float*)d_in[8];   // ff_norm_w
  const float* w_in   = (const float*)d_in[9];   // [4,5460,1024]
  const float* b_in   = (const float*)d_in[10];  // [4,5460]
  const float* w_out  = (const float*)d_in[11];  // [4,1024,2730]
  const float* b_out  = (const float*)d_in[12];  // [4,1024]
  const float* finw   = (const float*)d_in[13];  // [1024]

  constexpr int QL = 1024 * 1024, KVL = 2048 * 1024, WOL = 1024 * 1024;
  constexpr int GL = 16 * 1024, WINL = FI2c * 1024, WOUTLp = 1024 * FIpc;

  char* base = (char*)d_ws;
  size_t off = 0;
  auto alloc = [&](size_t bytes) {
    char* p = base + off;
    off += (bytes + 255) & ~(size_t)255;
    return (void*)p;
  };
  float*          xf    = (float*)alloc((size_t)Mrows * Dc * 4);
  unsigned short* hb    = (unsigned short*)alloc((size_t)Mrows * Dc * 2);
  float*          qf    = (float*)alloc((size_t)Mrows * 1024 * 4);
  float*          kvf   = (float*)alloc((size_t)Mrows * 2048 * 4);
  float*          mixb  = (float*)alloc((size_t)Mrows * Hc * 4);
  float*          gateb = (float*)alloc((size_t)Mrows * Hc * 4);
  unsigned short* qb    = (unsigned short*)alloc((size_t)Mrows * 1024 * 2);
  unsigned short* kbuf  = (unsigned short*)alloc((size_t)Mrows * 1024 * 2);
  unsigned short* vbuf  = (unsigned short*)alloc((size_t)Mrows * 1024 * 2);
  float*          vres  = (float*)alloc((size_t)Mrows * 1024 * 4);
  unsigned short* aout  = (unsigned short*)alloc((size_t)Mrows * 1024 * 2);
  float*          ub    = (float*)alloc((size_t)Mrows * FI2c * 4);
  unsigned short* ggb   = (unsigned short*)alloc((size_t)Mrows * FIpc * 2);
  unsigned short* wq_b   = (unsigned short*)alloc((size_t)4 * QL * 2);
  unsigned short* wkv_b  = (unsigned short*)alloc((size_t)4 * KVL * 2);
  unsigned short* wo_b   = (unsigned short*)alloc((size_t)4 * WOL * 2);
  unsigned short* wg_b   = (unsigned short*)alloc((size_t)4 * GL * 2);
  unsigned short* wmix_b = (unsigned short*)alloc((size_t)4 * GL * 2);
  unsigned short* win_b  = (unsigned short*)alloc((size_t)4 * WINL * 2);
  unsigned short* wout_b = (unsigned short*)alloc((size_t)4 * WOUTLp * 2);

  auto cast = [&](const float* s, unsigned short* d, int n) {
    cast_f32_bf16<<<2048, 256, 0, stream>>>(s, d, n);
  };
  cast(wq, wq_b, 4 * QL);      cast(wkv, wkv_b, 4 * KVL);
  cast(wo, wo_b, 4 * WOL);     cast(wg, wg_b, 4 * GL);
  cast(wmix, wmix_b, 4 * GL);  cast(w_in, win_b, 4 * WINL);
  cast_f32_bf16_pad<<<2048, 256, 0, stream>>>(w_out, wout_b, 4 * 1024, FIc,
                                              FIpc);

  hipMemcpyAsync(xf, x_in, (size_t)Mrows * Dc * 4, hipMemcpyDeviceToDevice,
                 stream);

  // template dispatch: 0 = plain, 1 = +residual (accum), bias variants
  auto gemm = [&](const unsigned short* A, int lda, const unsigned short* W,
                  int ldw, float* C, const float* bias, int M_, int N_, int K_,
                  bool accum) {
    dim3 g((N_ + 63) / 64, (M_ + 127) / 128);
    if (accum) {
      if (bias)
        gemm_bf16_wmma<1, 1><<<g, 256, 0, stream>>>(A, lda, W, ldw, C, bias,
                                                    M_, N_, K_);
      else
        gemm_bf16_wmma<1, 0><<<g, 256, 0, stream>>>(A, lda, W, ldw, C, bias,
                                                    M_, N_, K_);
    } else {
      if (bias)
        gemm_bf16_wmma<0, 1><<<g, 256, 0, stream>>>(A, lda, W, ldw, C, bias,
                                                    M_, N_, K_);
      else
        gemm_bf16_wmma<0, 0><<<g, 256, 0, stream>>>(A, lda, W, ldw, C, bias,
                                                    M_, N_, K_);
    }
  };

  for (int i = 0; i < 4; ++i) {
    // --- attention block ---
    rmsnorm2_bf16<<<Mrows, 256, 0, stream>>>(xf, naw + i * Dc, nullptr, hb);
    gemm(hb, 1024, wq_b + (size_t)i * QL, 1024, qf, nullptr, Mrows, 1024,
         1024, false);
    gemm(hb, 1024, wkv_b + (size_t)i * KVL, 1024, kvf, nullptr, Mrows, 2048,
         1024, false);
    gemm(hb, 1024, wg_b + (size_t)i * GL, 1024, gateb, nullptr, Mrows, Hc,
         1024, false);
    if (i > 0)
      gemm(hb, 1024, wmix_b + (size_t)i * GL, 1024, mixb, nullptr, Mrows, Hc,
           1024, false);
    prep_qkv<<<dim3(Nc, Hc, Bc), 64, 0, stream>>>(
        qf, kvf, (i > 0) ? mixb : nullptr, vres, qb, kbuf, vbuf, i == 0);
    flash_attn<<<dim3(Nc / 128, Hc, Bc), 256, 0, stream>>>(qb, kbuf, vbuf,
                                                           gateb, aout);
    gemm(aout, 1024, wo_b + (size_t)i * WOL, 1024, xf, nullptr, Mrows, 1024,
         1024, true);

    // --- feedforward block (wrapper norm + FF pre-norm fused) ---
    rmsnorm2_bf16<<<Mrows, 256, 0, stream>>>(xf, nfw + i * Dc, ffw + i * Dc,
                                             hb);
    gemm(hb, 1024, win_b + (size_t)i * WINL, 1024, ub,
         b_in + (size_t)i * FI2c, Mrows, FI2c, 1024, false);
    geglu_kernel<<<8192, 256, 0, stream>>>(ub, ggb, Mrows * FIpc);
    gemm(ggb, FIpc, wout_b + (size_t)i * WOUTLp, FIpc, xf,
         b_out + (size_t)i * Dc, Mrows, 1024, FIpc, true);
  }

  final_rms<<<Mrows, 256, 0, stream>>>(xf, finw, (float*)d_out);
}
